// Noisy_Weights_BNN_51436528337046
// MI455X (gfx1250) — matrix operand order repaired and checked by reading, not compiled, not run on hardware
//
#include <hip/hip_runtime.h>
#include <hip/hip_bf16.h>

// ---------------------------------------------------------------------------
// Types for CDNA5 WMMA (wave32): v_wmma_f32_16x16x32_bf16
// ---------------------------------------------------------------------------
typedef __bf16        v16bf __attribute__((ext_vector_type(16)));
typedef float         v8f   __attribute__((ext_vector_type(8)));
typedef unsigned int  u32x4 __attribute__((ext_vector_type(4)));

#define S_REP   16
#define MROWS   2048          // 32*64
#define D0      784
#define D0P     800           // padded to multiple of 32
#define D1      512
#define D2      10
#define D2P     16            // padded N for layer 2

// LDS tile row pitch in bf16 elements: 32 data + 8 pad = 80B rows.
// 80B = 20 banks stride -> 16 b128 reads per half-wave cover all 64 banks.
#define KP      40

// Block tile (8 waves = 4 along M x 2 along N, each wave owns 64x64)
#define BM      256
#define BN      128

// ---------------------------------------------------------------------------
// Workspace layout (bytes, all 256-aligned)
// ---------------------------------------------------------------------------
#define OFF_BATCH   ((size_t)0)                         // bf16 [2048][800]
#define OFF_W0T     ((size_t)3276800)                   // bf16 [512][800]   (W0^T, k-padded)
#define OFF_W1NT    ((size_t)4096000)                   // bf16 [16][512][512] ((W1+noise)^T)
#define OFF_W2T     ((size_t)12484608)                  // bf16 [16][512]    (W2^T, n-padded)
#define OFF_X1      ((size_t)12500992)                  // bf16 [2048][512]
#define OFF_X2      ((size_t)14598144)                  // bf16 [16][2048][512]

__device__ __forceinline__ float sigmoidf(float x) {
    return 1.0f / (1.0f + __expf(-x));
}

// ---------------------------------------------------------------------------
// CDNA5 async global->LDS copy (ASYNCcnt-tracked), per ISA 10.x / 15.18.3.
// ---------------------------------------------------------------------------
__device__ __forceinline__ void async_copy_b128(unsigned lds_off, const void* gptr) {
    unsigned long long ga = (unsigned long long)(size_t)gptr;
    asm volatile("global_load_async_to_lds_b128 %0, %1, off"
                 :: "v"(lds_off), "v"(ga)
                 : "memory");
}
__device__ __forceinline__ void wait_async0() {
    asm volatile("s_wait_asynccnt 0" ::: "memory");
}

// ---------------------------------------------------------------------------
// WMMA fragment helpers (ISA 7.12.2 layouts, wave32)
// ---------------------------------------------------------------------------
__device__ __forceinline__ v16bf frag_pair(const __bf16* p, int second_off_halfs) {
    union U { u32x4 v; __bf16 h[8]; } lo, hi;
    lo.v = *(const u32x4*)(p);
    hi.v = *(const u32x4*)(p + second_off_halfs);
    v16bf f;
#pragma unroll
    for (int i = 0; i < 8; ++i) { f[i] = lo.h[i]; f[8 + i] = hi.h[i]; }
    return f;
}

// A-matrix 16x32 bf16: lanes 0-15 -> row M=lane, K = {0..7, 16..23}
//                      lanes 16-31 -> row M=lane-16, K = {8..15, 24..31}
__device__ __forceinline__ v16bf load_a_frag_g(const __bf16* __restrict__ A,
                                               int lda, int row0, int k0) {
    const int lane = threadIdx.x & 31;
    const int r    = row0 + (lane & 15);
    const int kk   = k0 + ((lane >> 4) << 3);
    return frag_pair(A + (size_t)r * lda + kk, 16);
}

// B-matrix 32x16 bf16 from transposed BT[N][K]: same K split as A
__device__ __forceinline__ v16bf load_b_frag_g(const __bf16* __restrict__ BT,
                                               int ldb, int n0, int k0) {
    const int lane = threadIdx.x & 31;
    const int c    = n0 + (lane & 15);
    const int kk   = k0 + ((lane >> 4) << 3);
    return frag_pair(BT + (size_t)c * ldb + kk, 16);
}

__device__ __forceinline__ v8f wmma_bf16(v16bf a, v16bf b, v8f c) {
    return __builtin_amdgcn_wmma_f32_16x16x32_bf16(false, a, false, b,
                                                   (short)0, c, false, false);
}

// C/D 16x16 f32: VGPR v, lanes 0-15 -> (M=v, N=lane); lanes 16-31 -> (M=v+8, N=lane-16)
__device__ __forceinline__ void store_tile_sig_bf16(__bf16* __restrict__ C, int ldc,
                                                    int m0, int n0, v8f acc) {
    const int lane = threadIdx.x & 31;
    const int col  = n0 + (lane & 15);
    const int rb   = m0 + ((lane >> 4) << 3);
#pragma unroll
    for (int v = 0; v < 8; ++v)
        C[(size_t)(rb + v) * ldc + col] = (__bf16)sigmoidf(acc[v]);
}

// ---------------------------------------------------------------------------
// Prep kernels: f32 -> bf16 conversion / transpose / pad / noise-add
// ---------------------------------------------------------------------------
__global__ void prep_batch(const float* __restrict__ in, __bf16* __restrict__ out) {
    int idx = blockIdx.x * blockDim.x + threadIdx.x;           // 2048*800
    if (idx >= MROWS * D0P) return;
    int r = idx / D0P, k = idx % D0P;
    out[idx] = (k < D0) ? (__bf16)in[(size_t)r * D0 + k] : (__bf16)0.0f;
}

__global__ void prep_w0t(const float* __restrict__ W0, __bf16* __restrict__ out) {
    int idx = blockIdx.x * blockDim.x + threadIdx.x;           // 512*800
    if (idx >= D1 * D0P) return;
    int n = idx / D0P, k = idx % D0P;
    out[idx] = (k < D0) ? (__bf16)W0[(size_t)k * D1 + n] : (__bf16)0.0f;
}

__global__ void prep_w1nt(const float* __restrict__ W1, const float* __restrict__ noise,
                          __bf16* __restrict__ out) {
    int idx = blockIdx.x * blockDim.x + threadIdx.x;           // 16*512*512
    if (idx >= S_REP * D1 * D1) return;
    int s = idx / (D1 * D1), rem = idx % (D1 * D1);
    int n = rem / D1, k = rem % D1;
    float w = W1[(size_t)k * D1 + n];
    if (s != 0) w += noise[(size_t)s * D1 * D1 + (size_t)k * D1 + n];
    out[idx] = (__bf16)w;
}

__global__ void prep_w2t(const float* __restrict__ W2, __bf16* __restrict__ out) {
    int idx = blockIdx.x * blockDim.x + threadIdx.x;           // 16*512
    if (idx >= D2P * D1) return;
    int n = idx / D1, k = idx % D1;
    out[idx] = (n < D2) ? (__bf16)W2[(size_t)k * D2 + n] : (__bf16)0.0f;
}

// ---------------------------------------------------------------------------
// Main GEMM (layers 0 and 1): out = sigmoid(A @ B), B given transposed.
// Block 256 thr = 8 waves (4 M x 2 N), wave tile 64x64 -> block tile 256x128.
// Per wave per K-step(32): 16 WMMAs vs 16 ds_load_b128 (32 FLOP per LDS byte),
// one wait_asynccnt+barrier per 16 WMMAs.
// Double-buffered LDS staged with CDNA5 async global->LDS copies.
// ---------------------------------------------------------------------------
__global__ void __launch_bounds__(256)
gemm_bf16_lds(const __bf16* __restrict__ A, size_t sA, int lda,
              const __bf16* __restrict__ BT, size_t sB, int ldb,
              int K,
              __bf16* __restrict__ out, size_t sO, int ldo) {
    __shared__ __align__(16) __bf16 As[2][BM][KP];   // 40 KB
    __shared__ __align__(16) __bf16 Bs[2][BN][KP];   // 20 KB

    const int s   = blockIdx.z;
    const __bf16* Ag = A  + s * sA + (size_t)blockIdx.x * BM * lda;
    const __bf16* Bg = BT + s * sB + (size_t)blockIdx.y * BN * ldb;
    __bf16*       O  = out + s * sO;

    const int t    = threadIdx.x;
    const int lane = t & 31;
    const int wave = t >> 5;
    const int m0l  = (wave & 3) * 64;                 // tile-local row
    const int n0l  = (wave >> 2) * 64;                // tile-local col
    const int mg0  = blockIdx.x * BM + m0l;           // global row
    const int ng0  = blockIdx.y * BN + n0l;           // global col

    // --- async stage of one 32-wide K chunk into LDS buffer `buf` ---
    auto stage = [&](int buf, int k0) {
        // A tile: 256 rows x 64B = 1024 x 16B chunks, 4 per thread
#pragma unroll
        for (int i = 0; i < 4; ++i) {
            int q = t + i * 256;
            int r = q >> 2, c = q & 3;
            unsigned loff = (unsigned)(size_t)&As[buf][r][c * 8];
            async_copy_b128(loff, Ag + (size_t)r * lda + k0 + c * 8);
        }
        // B tile: 128 rows x 64B = 512 x 16B chunks, 2 per thread
#pragma unroll
        for (int i = 0; i < 2; ++i) {
            int q = t + i * 256;
            int r = q >> 2, c = q & 3;
            unsigned loff = (unsigned)(size_t)&Bs[buf][r][c * 8];
            async_copy_b128(loff, Bg + (size_t)r * ldb + k0 + c * 8);
        }
    };

    v8f acc[4][4];
#pragma unroll
    for (int i = 0; i < 4; ++i)
#pragma unroll
        for (int j = 0; j < 4; ++j) acc[i][j] = (v8f){};

    stage(0, 0);
    wait_async0();
    __syncthreads();

    const int KT  = K / 32;
    const int ar  = lane & 15;                 // fragment row within 16
    const int akk = (lane >> 4) << 3;          // A frag: +0 / +8
    const int bkk = (lane >> 4) << 3;          // B frag: +0 / +8

    for (int kt = 0; kt < KT; ++kt) {
        const int cur = kt & 1;
        if (kt + 1 < KT) stage(cur ^ 1, (kt + 1) * 32);

        // Fragments from LDS (ds_load_b128; 80B row pitch avoids bank conflicts)
        v16bf af[4], bf[4];
#pragma unroll
        for (int i = 0; i < 4; ++i)
            af[i] = frag_pair(&As[cur][m0l + 16 * i + ar][akk], 16);
#pragma unroll
        for (int j = 0; j < 4; ++j)
            bf[j] = frag_pair(&Bs[cur][n0l + 16 * j + ar][bkk], 16);

#pragma unroll
        for (int i = 0; i < 4; ++i)
#pragma unroll
            for (int j = 0; j < 4; ++j)
                acc[i][j] = wmma_bf16(af[i], bf[j], acc[i][j]);

        wait_async0();       // next chunk staged
        __syncthreads();     // all waves done reading `cur`, staged data visible
    }

#pragma unroll
    for (int i = 0; i < 4; ++i)
#pragma unroll
        for (int j = 0; j < 4; ++j)
            store_tile_sig_bf16(O, ldo, mg0 + 16 * i, ng0 + 16 * j, acc[i][j]);
}

// ---------------------------------------------------------------------------
// Layer 2: out[s] = sigmoid(X2[s] @ W2)  [2048 x 10], N padded to 16, K = 512.
// Tiny (0.3 GFLOP): direct-global fragments. Wave tile 32x16, grid (8,1,16).
// ---------------------------------------------------------------------------
__global__ void __launch_bounds__(256)
gemm_layer2(const __bf16* __restrict__ X2, const __bf16* __restrict__ W2T,
            float* __restrict__ out) {
    const int s    = blockIdx.z;
    const int wave = threadIdx.x >> 5;
    const int m0 = blockIdx.x * 256 + wave * 32;
    const __bf16* A = X2 + (size_t)s * MROWS * D1;
    float* O = out + (size_t)s * MROWS * D2;
    v8f acc0 = {}, acc1 = {};
    for (int k0 = 0; k0 < D1; k0 += 32) {
        v16bf a0 = load_a_frag_g(A, D1, m0,      k0);
        v16bf a1 = load_a_frag_g(A, D1, m0 + 16, k0);
        v16bf b  = load_b_frag_g(W2T, D1, 0, k0);
        acc0 = wmma_bf16(a0, b, acc0);
        acc1 = wmma_bf16(a1, b, acc1);
    }
    const int lane = threadIdx.x & 31;
    const int col  = lane & 15;
    if (col < D2) {
        const int rb0 = m0 + ((lane >> 4) << 3);
#pragma unroll
        for (int v = 0; v < 8; ++v) {
            O[(size_t)(rb0 + v) * D2 + col]      = sigmoidf(acc0[v]);
            O[(size_t)(rb0 + 16 + v) * D2 + col] = sigmoidf(acc1[v]);
        }
    }
}

// ---------------------------------------------------------------------------
// Host-side launcher
// ---------------------------------------------------------------------------
extern "C" void kernel_launch(void* const* d_in, const int* in_sizes, int n_in,
                              void* d_out, int out_size, void* d_ws, size_t ws_size,
                              hipStream_t stream) {
    const float* batch = (const float*)d_in[0];   // [32,64,784]
    const float* W0    = (const float*)d_in[1];   // [784,512]
    const float* W1    = (const float*)d_in[2];   // [512,512]
    const float* W2    = (const float*)d_in[3];   // [512,10]
    const float* noise = (const float*)d_in[4];   // [16,512,512]
    float* out = (float*)d_out;                   // [16,32,64,10]

    char* ws = (char*)d_ws;
    __bf16* batch_b = (__bf16*)(ws + OFF_BATCH);
    __bf16* w0t     = (__bf16*)(ws + OFF_W0T);
    __bf16* w1nt    = (__bf16*)(ws + OFF_W1NT);
    __bf16* w2t     = (__bf16*)(ws + OFF_W2T);
    __bf16* x1      = (__bf16*)(ws + OFF_X1);
    __bf16* x2      = (__bf16*)(ws + OFF_X2);

    // ---- prep (bf16 convert / transpose / pad / add noise) ----
    {
        int n = MROWS * D0P;
        prep_batch<<<(n + 255) / 256, 256, 0, stream>>>(batch, batch_b);
    }
    {
        int n = D1 * D0P;
        prep_w0t<<<(n + 255) / 256, 256, 0, stream>>>(W0, w0t);
    }
    {
        int n = S_REP * D1 * D1;
        prep_w1nt<<<(n + 255) / 256, 256, 0, stream>>>(W1, noise, w1nt);
    }
    {
        int n = D2P * D1;
        prep_w2t<<<(n + 255) / 256, 256, 0, stream>>>(W2, w2t);
    }

    // ---- layer 0 (computed ONCE; identical across replicas) ----
    gemm_bf16_lds<<<dim3(MROWS / BM, D1 / BN, 1), 256, 0, stream>>>(
        batch_b, 0, D0P, w0t, 0, D0P, D0P, x1, 0, D1);

    // ---- layer 1 (per-replica noisy weights) ----
    gemm_bf16_lds<<<dim3(MROWS / BM, D1 / BN, S_REP), 256, 0, stream>>>(
        x1, 0, D1, w1nt, (size_t)D1 * D1, D1, D1,
        x2, (size_t)MROWS * D1, D1);

    // ---- layer 2 (fused sigmoid, f32 output) ----
    gemm_layer2<<<dim3(MROWS / 256, 1, S_REP), 256, 0, stream>>>(x2, w2t, out);
}
